// RelPartialLearnableMultiHeadAttn_25022479466875
// MI455X (gfx1250) — matrix operand order repaired and checked by reading, not compiled, not run on hardware
//
#include <hip/hip_runtime.h>

typedef __bf16 bf16_t;
typedef __attribute__((ext_vector_type(16))) __bf16 v16bf;
typedef __attribute__((ext_vector_type(8)))  __bf16 v8bf;
typedef __attribute__((ext_vector_type(8)))  float  v8f;

#define QLEN   2048
#define BSZ    2
#define DMODEL 1024
#define NHEAD  16
#define DHEAD  64
#define HDIM   1024
#define SCL    0.125f      /* 1/sqrt(64) */
#define LN_EPS 1e-5f

static __device__ __forceinline__ bf16_t f2bf(float f) {
  union { float f; unsigned u; } v; v.f = f;
  unsigned r = (v.u + 0x7FFFu + ((v.u >> 16) & 1u)) >> 16;
  union { unsigned short s; bf16_t b; } o; o.s = (unsigned short)r;
  return o.b;
}

/* ---------------- elementwise f32 -> bf16 ---------------- */
__global__ void cvt_f32_bf16(const float* __restrict__ in, bf16_t* __restrict__ out, int n) {
  int i = blockIdx.x * 256 + threadIdx.x;
  if (i < n) out[i] = f2bf(in[i]);
}

/* ------------- LDS-tiled transpose + convert: in[R,C] f32 -> out[C,R] bf16 ------------- */
__global__ void transpose_cvt(const float* __restrict__ in, bf16_t* __restrict__ out, int R, int C) {
  __shared__ float t[32][33];
  int x = threadIdx.x, y = threadIdx.y;     // 32 x 8
  int bx = blockIdx.x, by = blockIdx.y;
  #pragma unroll
  for (int yy = 0; yy < 32; yy += 8) {
    t[y + yy][x] = in[(size_t)(by * 32 + y + yy) * C + bx * 32 + x];
  }
  __syncthreads();
  #pragma unroll
  for (int yy = 0; yy < 32; yy += 8) {
    out[(size_t)(bx * 32 + y + yy) * R + by * 32 + x] = f2bf(t[x][y + yy]);
  }
}

/* ---------------- bf16 GEMM: C[M,1024] = A[M,1024] * Bt[1024,1024]^T ----------------
 * A row-major [M,K], Bt is weights pre-transposed to [N,K].
 * Block: 256 thr = 8 waves; each wave a 32x32 accumulator block (2 A-frags x
 * 2 B-frags -> 4 WMMAs per k-step). Fragments for step k+1 are loaded BEFORE
 * the WMMAs of step k (register double-buffering) so the WMMA pipe never waits
 * on just-issued loads.
 * mode 0: Q  -> qw = C+bq+rwb, qr = C+bq+rrb   (layout [b,nh,i,dh] bf16)
 * mode 1: K  -> C+bk                            (layout [b,nh,j,dh] bf16)
 * mode 2: V  -> C+bv                            (layout [b,nh,dh,j] bf16, transposed)
 * mode 3: R  -> C+br                            (layout [nh,j,dh]  bf16)
 * mode 4: O  -> y = C+bo+residual               (f32 [M,1024])
 */
__global__ void __launch_bounds__(256) gemm_bf16_nt(
    const bf16_t* __restrict__ A, const bf16_t* __restrict__ B, int M,
    const float* __restrict__ bias, const float* __restrict__ aux1,
    const float* __restrict__ aux2, const float* __restrict__ wres,
    void* __restrict__ out1, void* __restrict__ out2, int mode)
{
  const int K = 1024;
  int lane = threadIdx.x & 31;
  int wave = threadIdx.x >> 5;
  int bidN = blockIdx.x & 7;           // 1024/128
  int bidM = blockIdx.x >> 3;
  int m0 = bidM * 64 + (wave & 1) * 32;
  int n0 = bidN * 128 + (wave >> 1) * 32;
  int nl = lane & 15, kg = lane >> 4, lh = lane >> 4;

  const bf16_t* ap0 = A + (size_t)(m0 + nl) * K + kg * 16;
  const bf16_t* ap1 = A + (size_t)(m0 + 16 + nl) * K + kg * 16;
  const bf16_t* bp0 = B + (size_t)(n0 + nl) * K + kg * 16;
  const bf16_t* bp1 = B + (size_t)(n0 + 16 + nl) * K + kg * 16;

  v8f c00, c01, c10, c11;
  #pragma unroll
  for (int t = 0; t < 8; ++t) { c00[t] = 0.f; c01[t] = 0.f; c10[t] = 0.f; c11[t] = 0.f; }

  auto loadA = [&](const bf16_t* p, int k) -> v16bf {
    v8bf lo = *(const v8bf*)(p + k);
    v8bf hi = *(const v8bf*)(p + k + 8);
    v16bf a;
    #pragma unroll
    for (int t = 0; t < 8; ++t) { a[t] = lo[t]; a[t + 8] = hi[t]; }
    return a;
  };

  /* prologue: fragments for k0 = 0 */
  v16bf a0 = loadA(ap0, 0);
  v16bf a1 = loadA(ap1, 0);
  v16bf b0 = *(const v16bf*)(bp0);
  v16bf b1 = *(const v16bf*)(bp1);

  #pragma unroll 4
  for (int k0 = 0; k0 < K; k0 += 32) {
    int kn = k0 + 32;
    v16bf na0 = a0, na1 = a1, nb0 = b0, nb1 = b1;
    if (kn < K) {            /* issue next-step loads BEFORE this step's WMMAs */
      na0 = loadA(ap0, kn);
      na1 = loadA(ap1, kn);
      nb0 = *(const v16bf*)(bp0 + kn);
      nb1 = *(const v16bf*)(bp1 + kn);
    }
    if (k0 + 256 < K) {      /* stream prefetch: global_prefetch_b8, speculative */
      __builtin_prefetch(ap0 + k0 + 256, 0, 1);
      __builtin_prefetch(ap1 + k0 + 256, 0, 1);
      __builtin_prefetch(bp0 + k0 + 256, 0, 1);
      __builtin_prefetch(bp1 + k0 + 256, 0, 1);
    }
    c00 = __builtin_amdgcn_wmma_f32_16x16x32_bf16(false, a0, false, b0, (short)0, c00, false, false);
    c01 = __builtin_amdgcn_wmma_f32_16x16x32_bf16(false, a0, false, b1, (short)0, c01, false, false);
    c10 = __builtin_amdgcn_wmma_f32_16x16x32_bf16(false, a1, false, b0, (short)0, c10, false, false);
    c11 = __builtin_amdgcn_wmma_f32_16x16x32_bf16(false, a1, false, b1, (short)0, c11, false, false);
    a0 = na0; a1 = na1; b0 = nb0; b1 = nb1;
  }

  auto epi = [&](v8f& c, int mm, int nn) {
    #pragma unroll
    for (int p = 0; p < 8; ++p) {
      int m = mm + p + lh * 8;
      int n = nn + nl;
      float val = c[p] + bias[n];
      if (mode == 0) {
        int i = m >> 1, b = m & 1, nh = n >> 6, dh = n & 63;
        size_t q = ((size_t)((b * NHEAD + nh) * QLEN + i) << 6) + dh;
        ((bf16_t*)out1)[q] = f2bf(val + aux1[n]);
        ((bf16_t*)out2)[q] = f2bf(val + aux2[n]);
      } else if (mode == 1) {
        int j = m >> 1, b = m & 1, nh = n >> 6, dh = n & 63;
        ((bf16_t*)out1)[((size_t)((b * NHEAD + nh) * QLEN + j) << 6) + dh] = f2bf(val);
      } else if (mode == 2) {
        int j = m >> 1, b = m & 1, nh = n >> 6, dh = n & 63;
        ((bf16_t*)out1)[((size_t)((b * NHEAD + nh) * DHEAD + dh)) * QLEN + j] = f2bf(val);
      } else if (mode == 3) {
        int nh = n >> 6, dh = n & 63;
        ((bf16_t*)out1)[((size_t)(nh * QLEN + m) << 6) + dh] = f2bf(val);
      } else {
        ((float*)out1)[(size_t)m * DMODEL + n] = val + wres[(size_t)m * DMODEL + n];
      }
    }
  };
  epi(c00, m0, n0);
  epi(c01, m0, n0 + 16);
  epi(c10, m0 + 16, n0);
  epi(c11, m0 + 16, n0 + 16);
}

/* ---------------- flash attention with Transformer-XL relative shift ----------------
 * grid: bsz * NHEAD * (QLEN/128) blocks; 8 waves/block; wave = 16-row i-tile.
 * Per 32-wide j step: 4 WMMA (AC) + 6 WMMA (BD band) + shuffle diag-extract
 * + online softmax + LDS bounce for P + 4 WMMA (P x V).
 */
__global__ void __launch_bounds__(256) attn_kernel(
    const bf16_t* __restrict__ qw, const bf16_t* __restrict__ qr,
    const bf16_t* __restrict__ kb, const bf16_t* __restrict__ vb,
    const bf16_t* __restrict__ rk, bf16_t* __restrict__ av)
{
  __shared__ __align__(64) bf16_t pbuf[8][16][32];
  int lane = threadIdx.x & 31;
  int wave = threadIdx.x >> 5;
  int bm = blockIdx.x & 15;
  int nh = (blockIdx.x >> 4) & 15;
  int b  = blockIdx.x >> 8;
  int it0 = bm * 128 + wave * 16;
  int nl = lane & 15, kg = lane >> 4, lh = lane >> 4;

  const bf16_t* qwp = qw + ((size_t)(b * NHEAD + nh) * QLEN << 6);
  const bf16_t* qrp = qr + ((size_t)(b * NHEAD + nh) * QLEN << 6);
  const bf16_t* kp  = kb + ((size_t)(b * NHEAD + nh) * QLEN << 6);
  const bf16_t* vp  = vb + ((size_t)((b * NHEAD + nh) * DHEAD)) * QLEN;
  const bf16_t* rp  = rk + ((size_t)nh * QLEN << 6);

  /* Q fragments: row i = it0+nl; two 32-wide k-chunks over d=64 */
  v16bf qwA[2], qrA[2];
  #pragma unroll
  for (int kc = 0; kc < 2; ++kc) {
    const bf16_t* p1 = qwp + (size_t)(it0 + nl) * DHEAD + kc * 32 + kg * 16;
    v8bf lo = *(const v8bf*)p1, hi = *(const v8bf*)(p1 + 8);
    #pragma unroll
    for (int t = 0; t < 8; ++t) { qwA[kc][t] = lo[t]; qwA[kc][t + 8] = hi[t]; }
    const bf16_t* p2 = qrp + (size_t)(it0 + nl) * DHEAD + kc * 32 + kg * 16;
    lo = *(const v8bf*)p2; hi = *(const v8bf*)(p2 + 8);
    #pragma unroll
    for (int t = 0; t < 8; ++t) { qrA[kc][t] = lo[t]; qrA[kc][t + 8] = hi[t]; }
  }

  v8f O[4];
  #pragma unroll
  for (int n = 0; n < 4; ++n)
    #pragma unroll
    for (int t = 0; t < 8; ++t) O[n][t] = 0.f;
  float rmax[8], rsum[8];
  #pragma unroll
  for (int p = 0; p < 8; ++p) { rmax[p] = -1e30f; rsum[p] = 0.f; }

  int jmax = it0 + 15;
  for (int j0 = 0; j0 <= jmax; j0 += 32) {
    if (j0 + 32 <= jmax) {  /* prefetch next j-step K/V rows */
      __builtin_prefetch(kp + (size_t)(j0 + 32 + nl) * DHEAD, 0, 1);
      __builtin_prefetch(kp + (size_t)(j0 + 48 + nl) * DHEAD, 0, 1);
      __builtin_prefetch(vp + (size_t)nl * QLEN + j0 + 32, 0, 1);
      __builtin_prefetch(vp + (size_t)(48 + nl) * QLEN + j0 + 32, 0, 1);
    }
    /* ---- AC = (q + r_w_bias) . k^T : two 16-col tiles ---- */
    v8f AC[2];
    #pragma unroll
    for (int t = 0; t < 2; ++t) {
      v8f acc;
      #pragma unroll
      for (int q = 0; q < 8; ++q) acc[q] = 0.f;
      const bf16_t* kptr = kp + (size_t)(j0 + 16 * t + nl) * DHEAD + kg * 16;
      v16bf b0 = *(const v16bf*)(kptr);
      v16bf b1 = *(const v16bf*)(kptr + 32);
      acc = __builtin_amdgcn_wmma_f32_16x16x32_bf16(false, qwA[0], false, b0, (short)0, acc, false, false);
      acc = __builtin_amdgcn_wmma_f32_16x16x32_bf16(false, qwA[1], false, b1, (short)0, acc, false, false);
      AC[t] = acc;
    }
    /* ---- BD band: rel_shift => BD[i,j] = qr[i].rk[N-1-i+j]; 48-row rk window ---- */
    int base = QLEN - 16 - it0 + j0;
    v8f D[3];
    #pragma unroll
    for (int t = 0; t < 3; ++t) {
      v8f acc;
      #pragma unroll
      for (int q = 0; q < 8; ++q) acc[q] = 0.f;
      int rrow = base + 16 * t + nl;
      rrow = rrow < 0 ? 0 : (rrow > QLEN - 1 ? QLEN - 1 : rrow);
      const bf16_t* rptr = rp + (size_t)rrow * DHEAD + kg * 16;
      v16bf b0 = *(const v16bf*)(rptr);
      v16bf b1 = *(const v16bf*)(rptr + 32);
      acc = __builtin_amdgcn_wmma_f32_16x16x32_bf16(false, qrA[0], false, b0, (short)0, acc, false, false);
      acc = __builtin_amdgcn_wmma_f32_16x16x32_bf16(false, qrA[1], false, b1, (short)0, acc, false, false);
      D[t] = acc;
    }
    /* ---- diagonal extract + scale + causal mask + online softmax ---- */
    float S[2][8], tmax[8];
    #pragma unroll
    for (int p = 0; p < 8; ++p) {
      int Mrow = p + lh * 8;
      int irow = it0 + Mrow;
      {
        int rr = 15 - Mrow + nl;                       /* 0..30 -> D0/D1 */
        int src = (lane & 16) | (rr & 15);
        float va = __shfl(D[0][p], src, 32);
        float vx = __shfl(D[1][p], src, 32);
        float s = SCL * (AC[0][p] + ((rr < 16) ? va : vx));
        if (j0 + nl > irow) s = -1e30f;
        S[0][p] = s;
      }
      {
        int rr = 31 - Mrow + nl;                       /* 16..46 -> D1/D2 */
        int src = (lane & 16) | (rr & 15);
        float va = __shfl(D[1][p], src, 32);
        float vx = __shfl(D[2][p], src, 32);
        float s = SCL * (AC[1][p] + ((rr < 32) ? va : vx));
        if (j0 + 16 + nl > irow) s = -1e30f;
        S[1][p] = s;
      }
      float t = fmaxf(S[0][p], S[1][p]);
      #pragma unroll
      for (int m2 = 1; m2 < 16; m2 <<= 1) t = fmaxf(t, __shfl_xor(t, m2, 32));
      tmax[p] = t;
    }
    #pragma unroll
    for (int p = 0; p < 8; ++p) {
      float nm = fmaxf(rmax[p], tmax[p]);
      float corr = __expf(rmax[p] - nm);
      rmax[p] = nm;
      float e0 = __expf(S[0][p] - nm);
      float e1 = __expf(S[1][p] - nm);
      S[0][p] = e0; S[1][p] = e1;
      float rs = e0 + e1;
      #pragma unroll
      for (int m2 = 1; m2 < 16; m2 <<= 1) rs += __shfl_xor(rs, m2, 32);
      rsum[p] = rsum[p] * corr + rs;
      #pragma unroll
      for (int n = 0; n < 4; ++n) O[n][p] *= corr;
    }
    /* ---- P: C-layout -> LDS -> A-fragment (per-wave region, LDS in-order) ---- */
    #pragma unroll
    for (int p = 0; p < 8; ++p) {
      int row = p + lh * 8;
      pbuf[wave][row][nl]      = f2bf(S[0][p]);
      pbuf[wave][row][16 + nl] = f2bf(S[1][p]);
    }
    v16bf pA;
    {
      const bf16_t* pr = &pbuf[wave][nl][kg * 16];
      v8bf lo = *(const v8bf*)pr;
      v8bf hi = *(const v8bf*)(pr + 8);
      #pragma unroll
      for (int t = 0; t < 8; ++t) { pA[t] = lo[t]; pA[t + 8] = hi[t]; }
    }
    /* ---- O += P x V ---- */
    #pragma unroll
    for (int n = 0; n < 4; ++n) {
      const bf16_t* vptr = vp + (size_t)(n * 16 + nl) * QLEN + j0 + kg * 16;
      v16bf bvf = *(const v16bf*)vptr;
      O[n] = __builtin_amdgcn_wmma_f32_16x16x32_bf16(false, pA, false, bvf, (short)0, O[n], false, false);
    }
  }
  /* ---- finalize: attn_vec[i, b, nh*64+d] bf16 ---- */
  #pragma unroll
  for (int p = 0; p < 8; ++p) {
    float inv = 1.f / rsum[p];
    int i = it0 + p + lh * 8;
    #pragma unroll
    for (int n = 0; n < 4; ++n) {
      int hcol = nh * 64 + n * 16 + nl;
      av[(size_t)(i * BSZ + b) * HDIM + hcol] = f2bf(O[n][p] * inv);
    }
  }
}

/* ---------------- row LayerNorm over d=1024, one block per row ---------------- */
__global__ void __launch_bounds__(256) layernorm_kernel(
    const float* __restrict__ y, const float* __restrict__ g,
    const float* __restrict__ bb, float* __restrict__ out)
{
  __shared__ float red[256];
  int row = blockIdx.x;
  const float* yr = y + (size_t)row * DMODEL;
  float v[4];
  float s = 0.f;
  #pragma unroll
  for (int t = 0; t < 4; ++t) { v[t] = yr[threadIdx.x + t * 256]; s += v[t]; }
  red[threadIdx.x] = s; __syncthreads();
  for (int off = 128; off > 0; off >>= 1) {
    if (threadIdx.x < off) red[threadIdx.x] += red[threadIdx.x + off];
    __syncthreads();
  }
  float mu = red[0] * (1.f / DMODEL);
  __syncthreads();
  float s2 = 0.f;
  #pragma unroll
  for (int t = 0; t < 4; ++t) { float d = v[t] - mu; s2 += d * d; }
  red[threadIdx.x] = s2; __syncthreads();
  for (int off = 128; off > 0; off >>= 1) {
    if (threadIdx.x < off) red[threadIdx.x] += red[threadIdx.x + off];
    __syncthreads();
  }
  float rstd = rsqrtf(red[0] * (1.f / DMODEL) + LN_EPS);
  #pragma unroll
  for (int t = 0; t < 4; ++t) {
    int n = threadIdx.x + t * 256;
    out[(size_t)row * DMODEL + n] = (v[t] - mu) * rstd * g[n] + bb[n];
  }
}

extern "C" void kernel_launch(void* const* d_in, const int* in_sizes, int n_in,
                              void* d_out, int out_size, void* d_ws, size_t ws_size,
                              hipStream_t stream)
{
  (void)in_sizes; (void)n_in; (void)out_size; (void)ws_size;
  const float* w   = (const float*)d_in[0];
  const float* r   = (const float*)d_in[1];
  const float* rwb = (const float*)d_in[2];
  const float* rrb = (const float*)d_in[3];
  const float* Wq  = (const float*)d_in[4];
  const float* bq  = (const float*)d_in[5];
  const float* Wk  = (const float*)d_in[6];
  const float* bk  = (const float*)d_in[7];
  const float* Wv  = (const float*)d_in[8];
  const float* bvv = (const float*)d_in[9];
  const float* Wr  = (const float*)d_in[10];
  const float* br  = (const float*)d_in[11];
  const float* Wo  = (const float*)d_in[12];
  const float* bo  = (const float*)d_in[13];
  const float* lng = (const float*)d_in[14];
  const float* lnb = (const float*)d_in[15];
  float* out = (float*)d_out;

  char* ws = (char*)d_ws;
  size_t off = 0;
  auto carve = [&](size_t bytes) -> void* {
    void* p = ws + off; off += (bytes + 255) & ~(size_t)255; return p;
  };
  const size_t MB4096 = (size_t)4096 * 1024;
  bf16_t* wbf = (bf16_t*)carve(MB4096 * 2);
  bf16_t* rbf = (bf16_t*)carve((size_t)2048 * 1024 * 2);
  bf16_t* WqT = (bf16_t*)carve((size_t)1024 * 1024 * 2);
  bf16_t* WkT = (bf16_t*)carve((size_t)1024 * 1024 * 2);
  bf16_t* WvT = (bf16_t*)carve((size_t)1024 * 1024 * 2);
  bf16_t* WrT = (bf16_t*)carve((size_t)1024 * 1024 * 2);
  bf16_t* WoT = (bf16_t*)carve((size_t)1024 * 1024 * 2);
  bf16_t* qwb = (bf16_t*)carve(MB4096 * 2);
  bf16_t* qrb = (bf16_t*)carve(MB4096 * 2);
  bf16_t* kbw = (bf16_t*)carve(MB4096 * 2);
  bf16_t* vbw = (bf16_t*)carve(MB4096 * 2);
  bf16_t* rkw = (bf16_t*)carve((size_t)2048 * 1024 * 2);
  bf16_t* avw = (bf16_t*)carve(MB4096 * 2);
  float*  yw  = (float*)carve(MB4096 * 4);

  /* stage 0: precision conversion + weight transposes */
  cvt_f32_bf16<<<(4096 * 1024) / 256, 256, 0, stream>>>(w, wbf, 4096 * 1024);
  cvt_f32_bf16<<<(2048 * 1024) / 256, 256, 0, stream>>>(r, rbf, 2048 * 1024);
  dim3 tb(32, 8), tg(32, 32);
  transpose_cvt<<<tg, tb, 0, stream>>>(Wq, WqT, 1024, 1024);
  transpose_cvt<<<tg, tb, 0, stream>>>(Wk, WkT, 1024, 1024);
  transpose_cvt<<<tg, tb, 0, stream>>>(Wv, WvT, 1024, 1024);
  transpose_cvt<<<tg, tb, 0, stream>>>(Wr, WrT, 1024, 1024);
  transpose_cvt<<<tg, tb, 0, stream>>>(Wo, WoT, 1024, 1024);

  /* stage 1: projections (WMMA GEMMs, fused bias + layout permutes)
     grid = (M/64) * (1024/128) blocks of 8 waves, each wave 32x32 out */
  gemm_bf16_nt<<<512, 256, 0, stream>>>(wbf, WqT, 4096, bq, rwb, rrb, nullptr, qwb, qrb, 0);
  gemm_bf16_nt<<<512, 256, 0, stream>>>(wbf, WkT, 4096, bk, nullptr, nullptr, nullptr, kbw, nullptr, 1);
  gemm_bf16_nt<<<512, 256, 0, stream>>>(wbf, WvT, 4096, bvv, nullptr, nullptr, nullptr, vbw, nullptr, 2);
  gemm_bf16_nt<<<256, 256, 0, stream>>>(rbf, WrT, 2048, br, nullptr, nullptr, nullptr, rkw, nullptr, 3);

  /* stage 2: flash attention with rel-shift (14 WMMAs per 16x32 score block) */
  attn_kernel<<<BSZ * NHEAD * (QLEN / 128), 256, 0, stream>>>(qwb, qrb, kbw, vbw, rkw, avw);

  /* stage 3: output projection (+bo +residual), then LayerNorm */
  gemm_bf16_nt<<<512, 256, 0, stream>>>(avw, WoT, 4096, bo, nullptr, nullptr, w, yw, nullptr, 4);
  layernorm_kernel<<<4096, 256, 0, stream>>>(yw, lng, lnb, out);
}
